// SwinTransformerBlock_34024730919194
// MI455X (gfx1250) — compile-verified, matrix operand order checked by reading, not run on hardware
//
#include <hip/hip_runtime.h>
#include <hip/hip_bf16.h>

#define HH 56
#define WWID 56
#define WSZ 7
#define SHIFT 3
#define HEADS 4
#define DIM 128
#define NTOK 49
#define NPAD 64
#define EPSV 1e-5f

typedef __attribute__((ext_vector_type(16))) __bf16        v16bf;
typedef __attribute__((ext_vector_type(8)))  float         v8f;
typedef __attribute__((ext_vector_type(4))) unsigned int   u32x4;
typedef __attribute__((ext_vector_type(2))) unsigned int   u32x2;
typedef __attribute__((ext_vector_type(4)))  float         f32x4;

__device__ __forceinline__ unsigned short f2bf(float f) {
  unsigned int x = __float_as_uint(f);
  x += 0x7FFFu + ((x >> 16) & 1u);            // round-to-nearest-even
  return (unsigned short)(x >> 16);
}

union FragU { v16bf b; u32x4 q[2]; };

// Load a 16x32 bf16 operand tile (A layout; B uses identical packing from a
// [N][K] row-major matrix). Per CDNA5 ISA: lanes 0-15 hold row(lane&15), K
// {0..7,16..23}; lanes 16-31 hold the same row, K {8..15,24..31}.
__device__ __forceinline__ v16bf load_frag(const unsigned short* base, int ld) {
  int lane = threadIdx.x & 31;
  int row  = lane & 15;
  int kh   = (lane >> 4) << 3;                 // 0 or 8
  const unsigned short* p = base + row * ld + kh;
  FragU u;
  u.q[0] = *(const u32x4*)(p);                 // K = kh .. kh+7
  u.q[1] = *(const u32x4*)(p + 16);            // K = kh+16 .. kh+23
  return u.b;
}

__device__ __forceinline__ v8f wmma_bf16(v16bf a, v16bf b, v8f c) {
  return __builtin_amdgcn_wmma_f32_16x16x32_bf16(false, a, false, b,
                                                 (short)0, c, false, false);
}

__device__ __forceinline__ int waveid() {
  return __builtin_amdgcn_readfirstlane((int)(threadIdx.x >> 5));
}

// ---------------------------------------------------------------- weight prep
__global__ void k_prep(const float* __restrict__ qkv_w, const float* __restrict__ proj_w,
                       const float* __restrict__ w1,    const float* __restrict__ w2,
                       unsigned short* __restrict__ WqkvT, unsigned short* __restrict__ projT,
                       unsigned short* __restrict__ w1T,   unsigned short* __restrict__ w2T) {
  int i = blockIdx.x * 256 + threadIdx.x;
  if (i < 49152) {                                  // qkv_w [128,384] -> [384,128]
    int n = i / 128, k = i - n * 128;
    WqkvT[i] = f2bf(qkv_w[k * 384 + n]);
  } else if (i < 65536) {                           // proj_w [128,128] -> [128,128]T
    int j = i - 49152; int n = j / 128, k = j - n * 128;
    projT[j] = f2bf(proj_w[k * 128 + n]);
  } else if (i < 131072) {                          // mlp_w1 [128,512] -> [512,128]
    int j = i - 65536; int n = j / 128, k = j - n * 128;
    w1T[j] = f2bf(w1[k * 512 + n]);
  } else if (i < 196608) {                          // mlp_w2 [512,128] -> [128,512]
    int j = i - 131072; int n = j / 512, k = j - n * 512;
    w2T[j] = f2bf(w2[k * 128 + n]);
  }
}

// -------------------------------------------- LN1 + cyclic shift + partition
__global__ void k_ln1_part(const float* __restrict__ x,
                           const float* __restrict__ nw, const float* __restrict__ nb,
                           unsigned short* __restrict__ xw) {
  int w = blockIdx.x;
  int b = w >> 6, wi = w & 63;
  int wh = wi >> 3, wc = wi & 7;
  int lane = threadIdx.x & 31, wave = waveid();
  int ch = lane * 4;
  for (int it = 0; it < 8; ++it) {
    int t = wave * 8 + it;
    unsigned short* dst = xw + (size_t)w * (NPAD * DIM) + t * DIM + ch;
    if (t < NTOK) {
      int r = t / 7, c = t - r * 7;
      int hs = (wh * 7 + r + SHIFT) % HH;
      int cs = (wc * 7 + c + SHIFT) % WWID;
      const float* src = x + ((size_t)b * (HH * WWID) + hs * WWID + cs) * DIM + ch;
      f32x4 v = *(const f32x4*)src;
      float s  = v.x + v.y + v.z + v.w;
      float sq = v.x * v.x + v.y * v.y + v.z * v.z + v.w * v.w;
      for (int mk = 16; mk >= 1; mk >>= 1) {
        s += __shfl_xor(s, mk, 32); sq += __shfl_xor(sq, mk, 32);
      }
      float mean = s * (1.0f / 128.0f);
      float var  = sq * (1.0f / 128.0f) - mean * mean;
      float rstd = rsqrtf(var + EPSV);
      unsigned short o0 = f2bf((v.x - mean) * rstd * nw[ch + 0] + nb[ch + 0]);
      unsigned short o1 = f2bf((v.y - mean) * rstd * nw[ch + 1] + nb[ch + 1]);
      unsigned short o2 = f2bf((v.z - mean) * rstd * nw[ch + 2] + nb[ch + 2]);
      unsigned short o3 = f2bf((v.w - mean) * rstd * nw[ch + 3] + nb[ch + 3]);
      u32x2 pk; pk.x = o0 | ((unsigned)o1 << 16); pk.y = o2 | ((unsigned)o3 << 16);
      *(u32x2*)dst = pk;
    } else {                                        // zero pad rows 49..63
      u32x2 z = {0u, 0u};
      *(u32x2*)dst = z;
    }
  }
}

// ------------------------------------------------------------- QKV projection
// blockIdx.x = w*4 + rt  (row tile moved into grid so no loop-invariant
// B-fragments exist to be hoisted/spilled).
__global__ void k_qkv(const unsigned short* __restrict__ xw,
                      const unsigned short* __restrict__ wT,
                      const float* __restrict__ qkv_b,
                      unsigned short* __restrict__ qb,
                      unsigned short* __restrict__ kb,
                      unsigned short* __restrict__ vb) {
  int w  = blockIdx.x >> 2;
  int rt = blockIdx.x & 3;
  int wave = waveid(), lane = threadIdx.x & 31;
  int nlo = lane & 15, mbase = (lane >> 4) << 3;
  const unsigned short* A = xw + (size_t)w * (NPAD * DIM) + rt * 16 * DIM;
  v8f acc[3] = {};
  for (int kk = 0; kk < 4; ++kk) {
    v16bf a = load_frag(A + kk * 32, DIM);
#pragma unroll
    for (int c = 0; c < 3; ++c) {
      v16bf bf_ = load_frag(wT + (size_t)(wave * 48 + c * 16) * DIM + kk * 32, DIM);
      acc[c] = wmma_bf16(a, bf_, acc[c]);
    }
  }
#pragma unroll
  for (int c = 0; c < 3; ++c) {
    int n0 = wave * 48 + c * 16;                   // scalar (wave-uniform)
    int which = n0 >> 7;                           // scalar select q/k/v
    int head  = (n0 >> 5) & 3;                     // scalar (16-wide group
    int hd0   = n0 & 31;                           //  never crosses 32-bdry)
    int n = n0 + nlo;
    size_t w4 = (size_t)(w * 4 + head);
    if (which == 0) {
#pragma unroll
      for (int i = 0; i < 8; ++i) {
        int m = rt * 16 + i + mbase;
        qb[w4 * 2048 + m * 32 + hd0 + nlo] =
            f2bf((acc[c][i] + qkv_b[n]) * 0.17677669529663687f);
      }
    } else if (which == 1) {
#pragma unroll
      for (int i = 0; i < 8; ++i) {
        int m = rt * 16 + i + mbase;
        kb[w4 * 2048 + m * 32 + hd0 + nlo] = f2bf(acc[c][i] + qkv_b[n]);
      }
    } else {
#pragma unroll
      for (int i = 0; i < 8; ++i) {
        int m = rt * 16 + i + mbase;
        vb[w4 * 2048 + (hd0 + nlo) * 64 + m] = f2bf(acc[c][i] + qkv_b[n]);  // V^T
      }
    }
  }
}

// ---------------------------------------- attention: QK^T + bias/mask + softmax + PV
__global__ void k_attn(const unsigned short* __restrict__ qb,
                       const unsigned short* __restrict__ kb,
                       const unsigned short* __restrict__ vb,
                       const float* __restrict__ rpb,
                       unsigned short* __restrict__ outw) {
  __shared__ __align__(16) unsigned short plds[4][16][64];
  int bi = blockIdx.x;
  int w = bi >> 2, head = bi & 3;
  int wi = w & 63, wh = wi >> 3, wc = wi & 7;
  int wavem = waveid(), lane = threadIdx.x & 31;
  int nlo = lane & 15, mbase = (lane >> 4) << 3;
  size_t base = (size_t)bi * 2048;

  v16bf aq = load_frag(qb + base + wavem * 16 * 32, 32);
  v8f s[4];
#pragma unroll
  for (int ct = 0; ct < 4; ++ct) {
    v16bf bk = load_frag(kb + base + ct * 16 * 32, 32);
    v8f z = {};
    s[ct] = wmma_bf16(aq, bk, z);
  }
#pragma unroll
  for (int i = 0; i < 8; ++i) {
    int ti = wavem * 16 + i + mbase;
    float vals[4]; float mx = -1e30f;
#pragma unroll
    for (int ct = 0; ct < 4; ++ct) {
      int tj = ct * 16 + nlo;
      float v = s[ct][i];
      if (tj < NTOK) {
        if (ti < NTOK) {
          int ri = ti / 7, ci = ti - ri * 7;
          int rj = tj / 7, cj = tj - rj * 7;
          int idx = (ri - rj + 6) * 13 + (ci - cj + 6);
          v += rpb[idx * HEADS + head];
          int hi = wh * 7 + ri, wxi = wc * 7 + ci;
          int hj = wh * 7 + rj, wxj = wc * 7 + cj;
          int li = (hi < 49 ? 0 : (hi < 53 ? 1 : 2)) * 3 + (wxi < 49 ? 0 : (wxi < 53 ? 1 : 2));
          int lj = (hj < 49 ? 0 : (hj < 53 ? 1 : 2)) * 3 + (wxj < 49 ? 0 : (wxj < 53 ? 1 : 2));
          if (li != lj) v -= 100.0f;
        }
      } else v = -1e30f;
      vals[ct] = v;
      mx = fmaxf(mx, v);
    }
    for (int mk = 1; mk <= 8; mk <<= 1) mx = fmaxf(mx, __shfl_xor(mx, mk, 32));
    float sum = 0.f;
#pragma unroll
    for (int ct = 0; ct < 4; ++ct) {
      int tj = ct * 16 + nlo;
      float e = (tj < NTOK) ? expf(vals[ct] - mx) : 0.0f;
      vals[ct] = e; sum += e;
    }
    for (int mk = 1; mk <= 8; mk <<= 1) sum += __shfl_xor(sum, mk, 32);
    float inv = 1.0f / sum;
#pragma unroll
    for (int ct = 0; ct < 4; ++ct)
      plds[wavem][i + mbase][ct * 16 + nlo] = f2bf(vals[ct] * inv);
  }
  // P[16x64] @ V[64x32]
  v8f o[2] = {};
#pragma unroll
  for (int kk = 0; kk < 2; ++kk) {
    v16bf ap = load_frag(&plds[wavem][0][0] + kk * 32, 64);
#pragma unroll
    for (int nt = 0; nt < 2; ++nt) {
      v16bf bv = load_frag(vb + base + nt * 16 * 64 + kk * 32, 64);
      o[nt] = wmma_bf16(ap, bv, o[nt]);
    }
  }
#pragma unroll
  for (int nt = 0; nt < 2; ++nt) {
#pragma unroll
    for (int i = 0; i < 8; ++i) {
      int ti = wavem * 16 + i + mbase;
      if (ti < NTOK) {
        int chn = head * 32 + nt * 16 + nlo;
        outw[((size_t)w * NPAD + ti) * DIM + chn] = f2bf(o[nt][i]);
      }
    }
  }
}

// ------------------------- proj GEMM + window reverse + unshift + residual add
__global__ void k_proj(const unsigned short* __restrict__ outw,
                       const unsigned short* __restrict__ pT,
                       const float* __restrict__ proj_b,
                       const float* __restrict__ x,
                       float* __restrict__ out) {
  int w = blockIdx.x;
  int b = w >> 6, wi = w & 63, wh = wi >> 3, wc = wi & 7;
  int wave = waveid(), lane = threadIdx.x & 31;
  int nlo = lane & 15, mbase = (lane >> 4) << 3;
  int rt = wave & 3, chalf = wave >> 2;
  const unsigned short* A = outw + (size_t)w * (NPAD * DIM) + rt * 16 * DIM;
  v8f acc[4] = {};
  for (int kk = 0; kk < 4; ++kk) {
    v16bf a = load_frag(A + kk * 32, DIM);
#pragma unroll
    for (int c = 0; c < 4; ++c) {
      int n0 = (chalf * 4 + c) * 16;
      v16bf bf_ = load_frag(pT + (size_t)n0 * DIM + kk * 32, DIM);
      acc[c] = wmma_bf16(a, bf_, acc[c]);
    }
  }
#pragma unroll
  for (int c = 0; c < 4; ++c) {
    int n = (chalf * 4 + c) * 16 + nlo;
#pragma unroll
    for (int i = 0; i < 8; ++i) {
      int m = rt * 16 + i + mbase;
      if (m < NTOK) {
        int r = m / 7, cc = m - r * 7;
        int hd_ = (wh * 7 + r + SHIFT) % HH;
        int wd  = (wc * 7 + cc + SHIFT) % WWID;
        size_t idx = ((size_t)b * (HH * WWID) + hd_ * WWID + wd) * DIM + n;
        out[idx] = x[idx] + acc[c][i] + proj_b[n];
      }
    }
  }
}

// ---------------------------------------------- LN2 + fc1 + GELU (bf16 out)
// rt fixed per wave (2 waves per row tile); B-frags depend on the ctg loop
// var so they cannot be hoisted; A-frags are reused across ctg.
__global__ void k_mlp1(const float* __restrict__ xres,
                       const float* __restrict__ nw, const float* __restrict__ nb,
                       const unsigned short* __restrict__ w1T,
                       const float* __restrict__ b1,
                       unsigned short* __restrict__ h1) {
  __shared__ __align__(16) unsigned short xl[64 * DIM];
  size_t t0 = (size_t)blockIdx.x * 64;
  int wave = waveid(), lane = threadIdx.x & 31;
  int ch = lane * 4;
  for (int it = 0; it < 8; ++it) {
    int tt = wave * 8 + it;
    const float* src = xres + (t0 + tt) * DIM + ch;
    f32x4 v = *(const f32x4*)src;
    float s  = v.x + v.y + v.z + v.w;
    float sq = v.x * v.x + v.y * v.y + v.z * v.z + v.w * v.w;
    for (int mk = 16; mk >= 1; mk >>= 1) {
      s += __shfl_xor(s, mk, 32); sq += __shfl_xor(sq, mk, 32);
    }
    float mean = s * (1.0f / 128.0f);
    float var  = sq * (1.0f / 128.0f) - mean * mean;
    float rstd = rsqrtf(var + EPSV);
    unsigned short o0 = f2bf((v.x - mean) * rstd * nw[ch + 0] + nb[ch + 0]);
    unsigned short o1 = f2bf((v.y - mean) * rstd * nw[ch + 1] + nb[ch + 1]);
    unsigned short o2 = f2bf((v.z - mean) * rstd * nw[ch + 2] + nb[ch + 2]);
    unsigned short o3 = f2bf((v.w - mean) * rstd * nw[ch + 3] + nb[ch + 3]);
    u32x2 pk; pk.x = o0 | ((unsigned)o1 << 16); pk.y = o2 | ((unsigned)o3 << 16);
    *(u32x2*)(xl + tt * DIM + ch) = pk;
  }
  __syncthreads();
  int nlo = lane & 15, mbase = (lane >> 4) << 3;
  int rt = wave & 3, half = wave >> 2;             // 16 col tiles per wave
  for (int ctg = 0; ctg < 4; ++ctg) {
    v8f acc[4] = {};
    for (int kk = 0; kk < 4; ++kk) {
      v16bf a = load_frag(xl + rt * 16 * DIM + kk * 32, DIM);
#pragma unroll
      for (int c = 0; c < 4; ++c) {
        int n0 = (half * 16 + ctg * 4 + c) * 16;
        v16bf bfr = load_frag(w1T + (size_t)n0 * DIM + kk * 32, DIM);
        acc[c] = wmma_bf16(a, bfr, acc[c]);
      }
    }
#pragma unroll
    for (int c = 0; c < 4; ++c) {
      int n = (half * 16 + ctg * 4 + c) * 16 + nlo;
#pragma unroll
      for (int i = 0; i < 8; ++i) {
        int m = rt * 16 + i + mbase;
        float v = acc[c][i] + b1[n];
        float g = 0.5f * v * (1.0f + erff(v * 0.70710678118654752f));  // exact GELU
        h1[(t0 + m) * 512 + n] = f2bf(g);
      }
    }
  }
}

// --------------------------------------------------------- fc2 + residual add
__global__ void k_mlp2(const unsigned short* __restrict__ h1,
                       const unsigned short* __restrict__ w2T,
                       const float* __restrict__ b2,
                       float* __restrict__ out) {
  size_t t0 = (size_t)blockIdx.x * 64;
  int wave = waveid(), lane = threadIdx.x & 31;
  int nlo = lane & 15, mbase = (lane >> 4) << 3;
  int rt = wave & 3, nh = wave >> 2;
  v8f acc[4] = {};
  for (int kk = 0; kk < 16; ++kk) {
    v16bf a = load_frag(h1 + (t0 + rt * 16) * 512 + kk * 32, 512);
#pragma unroll
    for (int c = 0; c < 4; ++c) {
      int n0 = (nh * 4 + c) * 16;
      v16bf bfr = load_frag(w2T + (size_t)n0 * 512 + kk * 32, 512);
      acc[c] = wmma_bf16(a, bfr, acc[c]);
    }
  }
#pragma unroll
  for (int c = 0; c < 4; ++c) {
    int n = (nh * 4 + c) * 16 + nlo;
#pragma unroll
    for (int i = 0; i < 8; ++i) {
      int m = rt * 16 + i + mbase;
      size_t idx = (t0 + m) * DIM + n;
      out[idx] = out[idx] + acc[c][i] + b2[n];
    }
  }
}

extern "C" void kernel_launch(void* const* d_in, const int* in_sizes, int n_in,
                              void* d_out, int out_size, void* d_ws, size_t ws_size,
                              hipStream_t stream) {
  (void)in_sizes; (void)n_in; (void)out_size; (void)ws_size;
  const float* x      = (const float*)d_in[0];
  const float* qkv_w  = (const float*)d_in[1];
  const float* qkv_b  = (const float*)d_in[2];
  const float* proj_w = (const float*)d_in[3];
  const float* proj_b = (const float*)d_in[4];
  const float* rpb    = (const float*)d_in[5];
  const float* n1w    = (const float*)d_in[6];
  const float* n1b    = (const float*)d_in[7];
  const float* n2w    = (const float*)d_in[8];
  const float* n2b    = (const float*)d_in[9];
  const float* w1     = (const float*)d_in[10];
  const float* b1     = (const float*)d_in[11];
  const float* w2     = (const float*)d_in[12];
  const float* b2     = (const float*)d_in[13];
  float* out = (float*)d_out;

  unsigned short* ws    = (unsigned short*)d_ws;
  unsigned short* WqkvT = ws;                     // 49152
  unsigned short* projT = ws + 49152;             // 16384
  unsigned short* w1T   = ws + 65536;             // 65536
  unsigned short* w2T   = ws + 131072;            // 65536
  unsigned short* xw    = ws + 196608;            // 4096*64*128
  unsigned short* qbuf  = xw   + 33554432;        // 4096*4*64*32
  unsigned short* kbuf  = qbuf + 33554432;
  unsigned short* vbuf  = kbuf + 33554432;        // stored transposed [32][64]
  unsigned short* outw  = vbuf + 33554432;        // 4096*64*128
  unsigned short* h1    = qbuf;                   // reuse q/k/v/outw region (dead by then)

  k_prep    <<<768,   256, 0, stream>>>(qkv_w, proj_w, w1, w2, WqkvT, projT, w1T, w2T);
  k_ln1_part<<<4096,  256, 0, stream>>>(x, n1w, n1b, xw);
  k_qkv     <<<16384, 256, 0, stream>>>(xw, WqkvT, qkv_b, qbuf, kbuf, vbuf);
  k_attn    <<<16384, 128, 0, stream>>>(qbuf, kbuf, vbuf, rpb, outw);
  k_proj    <<<4096,  256, 0, stream>>>(outw, projT, proj_b, x, out);
  k_mlp1    <<<3136,  256, 0, stream>>>(out, n2w, n2b, w1T, b1, h1);
  k_mlp2    <<<3136,  256, 0, stream>>>(h1, w2T, b2, out);
}
